// AttEncode_37306085933059
// MI455X (gfx1250) — compile-verified, matrix operand order checked by reading, not run on hardware
//
#include <hip/hip_runtime.h>
#include <hip/hip_bf16.h>
#include <math.h>

typedef __bf16 bf16_t;
typedef __attribute__((ext_vector_type(16))) bf16_t v16bf;
typedef __attribute__((ext_vector_type(8)))  float  v8f;

#define B_   32
#define L_   1024
#define D_   200
#define DP   256
#define M_   (B_ * L_)
#define STACK 6
#define HEAD  8
#define LNN  ((float)(L_ * D_))
#define EPS_ 1e-5f

// ---------- helpers ----------
__device__ __forceinline__ bf16_t f2bf(float f) {
  union { float f; unsigned u; } a; a.f = f;
  unsigned r = (a.u + 0x7FFFu + ((a.u >> 16) & 1u)) >> 16;  // round-nearest-even
  union { unsigned short s; bf16_t b; } o; o.s = (unsigned short)r;
  return o.b;
}

__device__ __forceinline__ v8f zero8() {
  v8f z;
#pragma unroll
  for (int i = 0; i < 8; ++i) z[i] = 0.f;
  return z;
}

// Load one 16-bit A/B fragment: lane holds two contiguous 8-element bf16 chunks
// (k .. k+7) and (k+16 .. k+23); caller bakes in the lane-half offset.
__device__ __forceinline__ v16bf load_frag(const bf16_t* p) {
  union { v16bf v; uint4 q[2]; } u;
  u.q[0] = *(const uint4*)(p);
  u.q[1] = *(const uint4*)(p + 16);
  return u.v;
}

__device__ __forceinline__ v8f wmma_bf16(v16bf a, v16bf b, v8f c) {
  return __builtin_amdgcn_wmma_f32_16x16x32_bf16(false, a, false, b, (short)0, c,
                                                 false, false);
}

// ---------- kernels ----------
__global__ void k_count_zeros(const int* __restrict__ x, int* __restrict__ off) {
  __shared__ int sh[256];
  int b = blockIdx.x, t = threadIdx.x, c = 0;
  for (int i = t; i < L_; i += 256) c += (x[b * L_ + i] == 0);
  sh[t] = c; __syncthreads();
  for (int s = 128; s > 0; s >>= 1) { if (t < s) sh[t] += sh[t + s]; __syncthreads(); }
  if (t == 0) off[b] = sh[0];
}

__global__ void k_embed(const int* __restrict__ x, const float* __restrict__ embed,
                        const int* __restrict__ off, float* __restrict__ hbuf,
                        bf16_t* __restrict__ hbf) {
  int bl = blockIdx.x;              // b*L + l
  int b = bl / L_, l = bl % L_;
  int e = threadIdx.x;              // 0..255 (pad zone >= 200 stays 0)
  float v = 0.f;
  if (e < D_) {
    int tok = x[bl];
    v = embed[(size_t)tok * D_ + e];
    int j = l - off[b];
    if (j >= 0) {
      float jf = (float)j;
      // jf / 1000^(jf/E)  ==  jf * exp(-jf*ln(1000)/E)
      float angle = jf * __expf(-jf * (6.9077552789821368f / (float)D_));
      v += (e & 1) ? __sinf(angle) : __cosf(angle);
    }
  }
  size_t idx = (size_t)bl * DP + e;
  hbuf[idx] = v;
  hbf[idx]  = f2bf(v);
}

__global__ void k_prep_weights(const float* __restrict__ w_qkv, const float* __restrict__ b_qkv,
                               const float* __restrict__ w_fuse, const float* __restrict__ b_fuse,
                               const float* __restrict__ w1, const float* __restrict__ b1,
                               const float* __restrict__ w2, const float* __restrict__ b2,
                               bf16_t* wqkvT, bf16_t* wfuseT, bf16_t* w1T, bf16_t* w2T,
                               float* bqp, float* bfp, float* b1p, float* b2p) {
  int n = blockIdx.x, k = threadIdx.x;           // wT[n][k] = w[k][n], zero-padded
  bool in = (n < D_) && (k < D_);
  size_t o = (size_t)n * DP + k;
  wqkvT[o] = f2bf(in ? w_qkv[(size_t)k * D_ + n] : 0.f);
  w1T[o]   = f2bf(in ? w1[(size_t)k * D_ + n]   : 0.f);
  w2T[o]   = f2bf(in ? w2[(size_t)k * D_ + n]   : 0.f);
  float s = 0.f;
  if (in) {
    // tile(c, head) @ w_fuse == c @ sum_i w_fuse[i*D + k][n]
    for (int i = 0; i < HEAD; ++i) s += w_fuse[(size_t)(i * D_ + k) * D_ + n];
  }
  wfuseT[o] = f2bf(in ? s : 0.f);
  if (n == 0) {
    bqp[k] = (k < D_) ? b_qkv[k] : 0.f;
    bfp[k] = (k < D_) ? b_fuse[k] : 0.f;
    b1p[k] = (k < D_) ? b1[k] : 0.f;
    b2p[k] = (k < D_) ? b2[k] : 0.f;
  }
}

// C[M,256] = A[M,256](bf16) @ WT'[256,256] + bias, epilogue per mode:
//   mode 0: out_bf = result (bf16)
//   mode 1: io_f32[idx] = result + io_f32[idx]   (fused residual, in place)
//   mode 2: out_bf = relu(result) (bf16)
__global__ void k_gemm(const bf16_t* __restrict__ A, const bf16_t* __restrict__ WT,
                       const float* __restrict__ bias, float* io, bf16_t* outb, int mode) {
  int lane = threadIdx.x & 31;
  int wid  = threadIdx.x >> 5;
  int job  = blockIdx.x * 8 + wid;      // 1024 blocks * 8 waves = 8192 jobs
  int mt = job >> 2, ng = job & 3;      // 2048 m-tiles x 4 n-groups (64 cols each)
  int m0 = mt * 16, n0 = ng * 64;
  int lrow = lane & 15;
  int lhi  = (lane >> 4) * 8;           // lane-half k offset
  const bf16_t* arow = A + (size_t)(m0 + lrow) * DP + lhi;

  v8f acc[4];
#pragma unroll
  for (int t = 0; t < 4; ++t) acc[t] = zero8();

#pragma unroll
  for (int kk = 0; kk < 8; ++kk) {      // K = 256 = 8 * 32
    v16bf af = load_frag(arow + kk * 32);
#pragma unroll
    for (int t = 0; t < 4; ++t) {
      const bf16_t* brow = WT + (size_t)(n0 + t * 16 + lrow) * DP + kk * 32 + lhi;
      acc[t] = wmma_bf16(af, load_frag(brow), acc[t]);
    }
  }

#pragma unroll
  for (int t = 0; t < 4; ++t) {
    int col = n0 + t * 16 + lrow;
    float bv = bias[col];
#pragma unroll
    for (int r = 0; r < 8; ++r) {
      int row = m0 + r + lhi;
      size_t idx = (size_t)row * DP + col;
      float v = acc[t][r] + bv;
      if (mode == 0)      outb[idx] = f2bf(v);
      else if (mode == 1) io[idx]   = v + io[idx];
      else                outb[idx] = f2bf(v > 0.f ? v : 0.f);
    }
  }
}

// qT[b][d][l] = qbf[b][l][d]   (32x32 LDS tiles)
__global__ void k_transpose(const bf16_t* __restrict__ qbf, bf16_t* __restrict__ qT) {
  __shared__ bf16_t tile[32][33];
  int blk = blockIdx.x;                 // B * 32 * 8 = 8192
  int b = blk >> 8;
  int rem = blk & 255;
  int tl = rem >> 3, td = rem & 7;
  int tx = threadIdx.x & 31, ty = threadIdx.x >> 5;   // 256 thr
  int l0 = tl * 32, d0 = td * 32;
#pragma unroll
  for (int i = 0; i < 4; ++i)
    tile[ty + 8 * i][tx] = qbf[(size_t)(b * L_ + l0 + ty + 8 * i) * DP + d0 + tx];
  __syncthreads();
#pragma unroll
  for (int i = 0; i < 4; ++i)
    qT[(size_t)(b * DP + d0 + ty + 8 * i) * L_ + l0 + tx] = tile[tx][ty + 8 * i];
}

// Self-attention with q=k=v. One wave owns 16 query rows + a 16x1024 f32 score
// strip in LDS (64KB); 4 waves/WG = 256KB LDS (<= 320KB/WGP). Phase 1: S=QK^T
// via WMMA into LDS. Phase 2: softmax in LDS, repack P as bf16 in place.
// Phase 3: O = P@V via WMMA (A-frags straight from LDS, B-frags from qT).
__global__ void k_attention(const bf16_t* __restrict__ q, const bf16_t* __restrict__ qT,
                            bf16_t* __restrict__ cbf) {
  extern __shared__ float smem[];       // 4 * 16 * 1024 floats
  int lane = threadIdx.x & 31;
  int wid  = threadIdx.x >> 5;          // 0..3
  int b    = blockIdx.x >> 4;           // 512 blocks
  int m0   = (blockIdx.x & 15) * 64 + wid * 16;
  int lrow = lane & 15;
  int lhi  = (lane >> 4) * 8;
  const float scale = 0.07071067811865475f;   // 1/sqrt(200)
  float* sbase = smem + (size_t)wid * 16 * 1024;

  // Q fragments for this 16-row tile, all 8 k-steps resident in VGPRs.
  const bf16_t* qrow = q + (size_t)(b * L_ + m0 + lrow) * DP + lhi;
  v16bf qf[8];
#pragma unroll
  for (int kk = 0; kk < 8; ++kk) qf[kk] = load_frag(qrow + kk * 32);

  // ---- Phase 1: scores -> LDS ----
  for (int nt = 0; nt < 64; ++nt) {
    int n0 = nt * 16;
    const bf16_t* krow = q + (size_t)(b * L_ + n0 + lrow) * DP + lhi;
    v8f acc = zero8();
#pragma unroll
    for (int kk = 0; kk < 8; ++kk)
      acc = wmma_bf16(qf[kk], load_frag(krow + kk * 32), acc);
    int col = n0 + lrow;                // C layout: lane=col, vgpr r = row (+8 hi half)
#pragma unroll
    for (int r = 0; r < 8; ++r)
      sbase[(r + lhi) * 1024 + col] = acc[r] * scale;
  }

  // ---- Phase 2: softmax (lanes 0..15, one row each; same-wave LDS is in-order) ----
  if (lane < 16) {
    float* srow = sbase + lane * 1024;
    float mx = -3.0e38f;
    for (int k = 0; k < 1024; ++k) mx = fmaxf(mx, srow[k]);
    float sum = 0.f;
    for (int k = 0; k < 1024; ++k) { float e = __expf(srow[k] - mx); srow[k] = e; sum += e; }
    float inv = 1.f / sum;
    bf16_t* prow = (bf16_t*)srow;       // repack bf16 into front half of the row
    for (int k = 0; k < 1024; ++k) prow[k] = f2bf(srow[k] * inv);
  }

  // ---- Phase 3: O = P @ V ----
  const char* pbytes = (const char*)sbase;   // row m at byte offset m*4096
  for (int jt = 0; jt < 16; ++jt) {
    int n0 = jt * 16;
    v8f acc = zero8();
    for (int t = 0; t < 32; ++t) {      // 1024 keys = 32 * 32
      const bf16_t* pf = (const bf16_t*)(pbytes + (size_t)lrow * 4096) + t * 32 + lhi;
      const bf16_t* vf = qT + (size_t)(b * DP + n0 + lrow) * L_ + t * 32 + lhi;
      acc = wmma_bf16(load_frag(pf), load_frag(vf), acc);
    }
    int col = n0 + lrow;
#pragma unroll
    for (int r = 0; r < 8; ++r)
      cbf[(size_t)(b * L_ + m0 + r + lhi) * DP + col] = f2bf(acc[r]);
  }
}

__global__ void k_zero_red(float* red) { if (threadIdx.x < 64) red[threadIdx.x] = 0.f; }

__global__ void k_ln_sum(const float* __restrict__ hbuf, float* __restrict__ red) {
  __shared__ float sh[256], sh2[256];
  int blk = blockIdx.x;                 // B * 128, each block = 8 rows
  int b = blk >> 7;
  int r0 = (blk & 127) * 8;
  int t = threadIdx.x;
  float s = 0.f, ss = 0.f;
  for (int i = t; i < 8 * D_; i += 256) {
    int row = r0 + i / D_, e = i % D_;
    float v = hbuf[(size_t)(b * L_ + row) * DP + e];
    s += v; ss += v * v;
  }
  sh[t] = s; sh2[t] = ss; __syncthreads();
  for (int st = 128; st > 0; st >>= 1) {
    if (t < st) { sh[t] += sh[t + st]; sh2[t] += sh2[t + st]; }
    __syncthreads();
  }
  if (t == 0) { atomicAdd(&red[b * 2], sh[0]); atomicAdd(&red[b * 2 + 1], sh2[0]); }
}

__global__ void k_ln_norm(float* __restrict__ hbuf, bf16_t* __restrict__ hbf,
                          const float* __restrict__ red) {
  int bl = blockIdx.x;
  int b = bl / L_;
  int e = threadIdx.x;
  float mu  = red[b * 2] / LNN;
  float var = red[b * 2 + 1] / LNN - mu * mu;
  float inv = rsqrtf(var + EPS_);
  size_t idx = (size_t)bl * DP + e;
  float v = (e < D_) ? (hbuf[idx] - mu) * inv : 0.f;   // keep pad zone at zero
  hbuf[idx] = v;
  hbf[idx]  = f2bf(v);
}

__global__ void k_pack(const float* __restrict__ hbuf, float* __restrict__ out) {
  int bl = blockIdx.x, e = threadIdx.x;
  if (e < D_) out[(size_t)bl * D_ + e] = hbuf[(size_t)bl * DP + e];
}

// ---------- launch ----------
extern "C" void kernel_launch(void* const* d_in, const int* in_sizes, int n_in,
                              void* d_out, int out_size, void* d_ws, size_t ws_size,
                              hipStream_t stream) {
  const int*   x      = (const int*)d_in[0];
  const float* embed  = (const float*)d_in[1];
  const float* w_qkv  = (const float*)d_in[2];
  const float* b_qkv  = (const float*)d_in[3];
  const float* w_fuse = (const float*)d_in[4];
  const float* b_fuse = (const float*)d_in[5];
  const float* w1     = (const float*)d_in[6];
  const float* b1     = (const float*)d_in[7];
  const float* w2     = (const float*)d_in[8];
  const float* b2     = (const float*)d_in[9];
  // d_in[10]=stack(6), d_in[11]=head(8): fixed by the reference, compiled in.

  char* w = (char*)d_ws;
  auto take = [&](size_t bytes) {
    char* p = w; w += (bytes + 255) & ~(size_t)255; return p;
  };
  float*  hbuf  = (float*) take((size_t)M_ * DP * 4);   // 33.5 MB current h (f32, padded)
  bf16_t* hbf   = (bf16_t*)take((size_t)M_ * DP * 2);   // bf16 mirror of h (GEMM A input)
  bf16_t* qbf   = (bf16_t*)take((size_t)M_ * DP * 2);   // q (bf16)
  bf16_t* qT    = (bf16_t*)take((size_t)B_ * DP * L_ * 2); // q transposed (V for P@V)
  bf16_t* cbf   = (bf16_t*)take((size_t)M_ * DP * 2);   // attn out / ffn mid (bf16)
  bf16_t* wqkvT = (bf16_t*)take((size_t)DP * DP * 2);
  bf16_t* wfuseT= (bf16_t*)take((size_t)DP * DP * 2);
  bf16_t* w1T   = (bf16_t*)take((size_t)DP * DP * 2);
  bf16_t* w2T   = (bf16_t*)take((size_t)DP * DP * 2);
  float*  bqp   = (float*) take(DP * 4);
  float*  bfp   = (float*) take(DP * 4);
  float*  b1p   = (float*) take(DP * 4);
  float*  b2p   = (float*) take(DP * 4);
  int*    off   = (int*)   take(B_ * 4);
  float*  red   = (float*) take(64 * 4);

  k_count_zeros<<<B_, 256, 0, stream>>>(x, off);
  k_embed<<<M_, 256, 0, stream>>>(x, embed, off, hbuf, hbf);
  k_prep_weights<<<DP, 256, 0, stream>>>(w_qkv, b_qkv, w_fuse, b_fuse, w1, b1, w2, b2,
                                         wqkvT, wfuseT, w1T, w2T, bqp, bfp, b1p, b2p);

  for (int s = 0; s < STACK; ++s) {
    // q = h @ Wqkv + b   (bf16 out)
    k_gemm<<<1024, 256, 0, stream>>>(hbf, wqkvT, bqp, nullptr, qbf, 0);
    k_transpose<<<B_ * 256, 256, 0, stream>>>(qbf, qT);
    // c = softmax(q q^T / sqrt(200)) q   (LDS-resident scores, 256KB/WG)
    k_attention<<<B_ * 16, 128, 4 * 16 * 1024 * sizeof(float), stream>>>(qbf, qT, cbf);
    // h = c @ Wfuse_eff + b + r   (fused residual, in place)
    k_gemm<<<1024, 256, 0, stream>>>(cbf, wfuseT, bfp, hbuf, nullptr, 1);
    k_zero_red<<<1, 64, 0, stream>>>(red);
    k_ln_sum<<<B_ * 128, 256, 0, stream>>>(hbuf, red);
    k_ln_norm<<<M_, 256, 0, stream>>>(hbuf, hbf, red);
    // ffn: t = relu(h@W1+b1); h = t@W2+b2 + r; ln2
    k_gemm<<<1024, 256, 0, stream>>>(hbf, w1T, b1p, nullptr, cbf, 2);
    k_gemm<<<1024, 256, 0, stream>>>(cbf, w2T, b2p, hbuf, nullptr, 1);
    k_zero_red<<<1, 64, 0, stream>>>(red);
    k_ln_sum<<<B_ * 128, 256, 0, stream>>>(hbuf, red);
    k_ln_norm<<<M_, 256, 0, stream>>>(hbuf, hbf, red);
  }
  k_pack<<<M_, 256, 0, stream>>>(hbuf, (float*)d_out);
}
